// DeepGraphSurv_49091476193810
// MI455X (gfx1250) — compile-verified, matrix operand order checked by reading, not run on hardware
//
#include <hip/hip_runtime.h>
#include <hip/hip_bf16.h>

typedef __attribute__((ext_vector_type(16))) __bf16 v16bf;
typedef __attribute__((ext_vector_type(8)))  __bf16 v8bf;
typedef __attribute__((ext_vector_type(8)))  float  v8f;

// ---------------------------------------------------------------------------
// fp32 -> bf16 flat conversion
// ---------------------------------------------------------------------------
__global__ void cvt_bf16_kernel(const float* __restrict__ in, __bf16* __restrict__ out, size_t n) {
    size_t i = (size_t)blockIdx.x * blockDim.x + threadIdx.x;
    if (i < n) out[i] = (__bf16)in[i];
}

// ---------------------------------------------------------------------------
// f32 (R x C, batched) -> optional bf16 row-major copy + bf16 transposed copy.
// LDS-tiled 32x32 transpose; block (32,8); grid (C/32, R/32, batch).
// ---------------------------------------------------------------------------
__global__ void pack_transpose_kernel(const float* __restrict__ in,
                                      __bf16* __restrict__ out_rm,
                                      __bf16* __restrict__ out_tr,
                                      int R, int C) {
    __shared__ float tile[32][33];
    const size_t bofs = (size_t)blockIdx.z * R * C;
    in += bofs;
    if (out_rm) out_rm += bofs;
    out_tr += bofs;
    const int r0 = blockIdx.y * 32, c0 = blockIdx.x * 32;
    const int tx = threadIdx.x, ty = threadIdx.y;
    #pragma unroll
    for (int i = 0; i < 32; i += 8) {
        float v = in[(size_t)(r0 + ty + i) * C + (c0 + tx)];
        tile[ty + i][tx] = v;
        if (out_rm) out_rm[(size_t)(r0 + ty + i) * C + (c0 + tx)] = (__bf16)v;
    }
    __syncthreads();
    #pragma unroll
    for (int i = 0; i < 32; i += 8)
        out_tr[(size_t)(c0 + ty + i) * R + (r0 + tx)] = (__bf16)tile[tx][ty + i];
}

// ---------------------------------------------------------------------------
// WMMA bf16 GEMM:  C(f32, MxN) = epilogue( A(bf16, MxK row-major)
//                                        x Bt(bf16, NxK row-major)^T )
// epilogue: + residual(f32 MxN) + bias(f32 N), ReLU, * mask(f32 M)
// Block: 256 threads = 8 waves; block tile 64x128; wave tile 32x32.
// Fragment loads per the CDNA5 16-bit A layout: lane half h holds
// K in [8h,8h+8) and [8h+16,8h+24)  => two aligned b128 loads per fragment.
// ---------------------------------------------------------------------------
__device__ __forceinline__ v16bf load_frag(const __bf16* p) {
    v8bf lo = *(const v8bf*)(p);
    v8bf hi = *(const v8bf*)(p + 16);
    return __builtin_shufflevector(lo, hi, 0, 1, 2, 3, 4, 5, 6, 7,
                                           8, 9, 10, 11, 12, 13, 14, 15);
}

__device__ __forceinline__ void epilogue_tile(v8f acc, int rowBase, int col, int h,
                                              float* __restrict__ C,
                                              const float* __restrict__ Res,
                                              const float* __restrict__ bias,
                                              const float* __restrict__ mask,
                                              int N, int relu) {
    #pragma unroll
    for (int v = 0; v < 8; ++v) {
        int row = rowBase + v + 8 * h;
        float c = acc[v];
        if (Res)  c += Res[(size_t)row * N + col];
        if (bias) c += bias[col];
        if (relu) c = fmaxf(c, 0.0f);
        if (mask) c *= mask[row];
        C[(size_t)row * N + col] = c;
    }
}

__global__ __launch_bounds__(256) void gemm_wmma_bf16(
    const __bf16* __restrict__ A,  long long sA,
    const __bf16* __restrict__ Bt, long long sB,
    float* __restrict__ C,         long long sC,
    const float* __restrict__ Res, long long sR,
    const float* __restrict__ bias,
    const float* __restrict__ mask, long long sM,
    int M, int N, int K, int relu) {
    const int bz = blockIdx.z;
    A  += (size_t)bz * sA;
    Bt += (size_t)bz * sB;
    C  += (size_t)bz * sC;
    if (Res)  Res  += (size_t)bz * sR;
    if (mask) mask += (size_t)bz * sM;

    const int w = threadIdx.x >> 5, lane = threadIdx.x & 31;
    const int h = lane >> 4, r16 = lane & 15;
    const int wm = w & 1, wn = w >> 1;
    const int tM = blockIdx.y * 64 + wm * 32;   // wave row base
    const int tN = blockIdx.x * 128 + wn * 32;  // wave col base

    const __bf16* a0 = A  + (size_t)(tM + r16) * K + 8 * h;
    const __bf16* a1 = a0 + (size_t)16 * K;
    const __bf16* b0 = Bt + (size_t)(tN + r16) * K + 8 * h;
    const __bf16* b1 = b0 + (size_t)16 * K;

    v8f acc00 = {}, acc01 = {}, acc10 = {}, acc11 = {};
    for (int k = 0; k < K; k += 32) {
        v16bf fa0 = load_frag(a0 + k);
        v16bf fa1 = load_frag(a1 + k);
        v16bf fb0 = load_frag(b0 + k);
        v16bf fb1 = load_frag(b1 + k);
        __builtin_prefetch(a0 + k + 128, 0, 0);
        __builtin_prefetch(a1 + k + 128, 0, 0);
        __builtin_prefetch(b0 + k + 128, 0, 0);
        __builtin_prefetch(b1 + k + 128, 0, 0);
        acc00 = __builtin_amdgcn_wmma_f32_16x16x32_bf16(false, fa0, false, fb0, (short)0, acc00, false, false);
        acc01 = __builtin_amdgcn_wmma_f32_16x16x32_bf16(false, fa0, false, fb1, (short)0, acc01, false, false);
        acc10 = __builtin_amdgcn_wmma_f32_16x16x32_bf16(false, fa1, false, fb0, (short)0, acc10, false, false);
        acc11 = __builtin_amdgcn_wmma_f32_16x16x32_bf16(false, fa1, false, fb1, (short)0, acc11, false, false);
    }
    epilogue_tile(acc00, tM,      tN      + r16, h, C, Res, bias, mask, N, relu);
    epilogue_tile(acc01, tM,      tN + 16 + r16, h, C, Res, bias, mask, N, relu);
    epilogue_tile(acc10, tM + 16, tN      + r16, h, C, Res, bias, mask, N, relu);
    epilogue_tile(acc11, tM + 16, tN + 16 + r16, h, C, Res, bias, mask, N, relu);
}

// ---------------------------------------------------------------------------
// att[b,n] = dot(Hh1[b,n,:], W2) + b2
// ---------------------------------------------------------------------------
__global__ void att_kernel(const float* __restrict__ Hh1, const float* __restrict__ W2,
                           const float* __restrict__ b2, float* __restrict__ att,
                           int total, int Adim) {
    int idx = blockIdx.x * blockDim.x + threadIdx.x;
    if (idx >= total) return;
    const float* r = Hh1 + (size_t)idx * Adim;
    float s = 0.0f;
    for (int a = 0; a < Adim; ++a) s += r[a] * W2[a];
    att[idx] = s + b2[0];
}

// ---------------------------------------------------------------------------
// softmax over nodes (axis=1), one block per batch
// ---------------------------------------------------------------------------
__global__ void softmax_kernel(const float* __restrict__ att, float* __restrict__ attn, int NN) {
    __shared__ float red[256];
    const int b = blockIdx.x, tid = threadIdx.x;
    const float* x = att + (size_t)b * NN;
    float mx = -3.4e38f;
    for (int i = tid; i < NN; i += 256) mx = fmaxf(mx, x[i]);
    red[tid] = mx; __syncthreads();
    for (int s = 128; s > 0; s >>= 1) { if (tid < s) red[tid] = fmaxf(red[tid], red[tid + s]); __syncthreads(); }
    mx = red[0]; __syncthreads();
    float sum = 0.0f;
    for (int i = tid; i < NN; i += 256) sum += __expf(x[i] - mx);
    red[tid] = sum; __syncthreads();
    for (int s = 128; s > 0; s >>= 1) { if (tid < s) red[tid] += red[tid + s]; __syncthreads(); }
    float inv = 1.0f / red[0];
    for (int i = tid; i < NN; i += 256) attn[(size_t)b * NN + i] = __expf(x[i] - mx) * inv;
}

// ---------------------------------------------------------------------------
// z[b,h] = sum_n Y[b,n,h] * attn[b,n]; grid (B, H/256)
// ---------------------------------------------------------------------------
__global__ void pool_kernel(const float* __restrict__ Y, const float* __restrict__ attn,
                            float* __restrict__ z, int NN, int H) {
    const int b = blockIdx.x;
    const int hh = blockIdx.y * blockDim.x + threadIdx.x;
    const float* Yb = Y + (size_t)b * NN * H;
    const float* ab = attn + (size_t)b * NN;
    float s = 0.0f;
    for (int n = 0; n < NN; ++n) s += Yb[(size_t)n * H + hh] * ab[n];
    z[b * H + hh] = s;
}

// ---------------------------------------------------------------------------
// logits[b] = dot(z[b,:], W_cls) + b_cls; one block per batch
// ---------------------------------------------------------------------------
__global__ void logits_kernel(const float* __restrict__ z, const float* __restrict__ Wc,
                              const float* __restrict__ bc, float* __restrict__ out, int H) {
    __shared__ float red[256];
    const int b = blockIdx.x, tid = threadIdx.x;
    float s = 0.0f;
    for (int i = tid; i < H; i += 256) s += z[b * H + i] * Wc[i];
    red[tid] = s; __syncthreads();
    for (int st = 128; st > 0; st >>= 1) { if (tid < st) red[tid] += red[tid + st]; __syncthreads(); }
    if (tid == 0) out[b] = red[0] + bc[0];
}

// ---------------------------------------------------------------------------
static inline char* carve(char*& p, size_t bytes) {
    char* r = p;
    p += (bytes + 255) & ~(size_t)255;
    return r;
}

extern "C" void kernel_launch(void* const* d_in, const int* in_sizes, int n_in,
                              void* d_out, int out_size, void* d_ws, size_t ws_size,
                              hipStream_t stream) {
    (void)in_sizes; (void)n_in; (void)out_size; (void)ws_size;
    const int B = 8, N = 2048, Din = 1024, H = 512, Adim = 128;

    const float* X      = (const float*)d_in[0];
    const float* adj    = (const float*)d_in[1];
    const float* mask   = (const float*)d_in[2];
    const float* W_fc   = (const float*)d_in[3];
    const float* b_fc   = (const float*)d_in[4];
    const float* W_att1 = (const float*)d_in[5];
    const float* b_att1 = (const float*)d_in[6];
    const float* W_att2 = (const float*)d_in[7];
    const float* b_att2 = (const float*)d_in[8];
    const float* W_cls  = (const float*)d_in[9];
    const float* b_cls  = (const float*)d_in[10];
    float* out = (float*)d_out;

    char* p = (char*)d_ws;
    __bf16* adjB = (__bf16*)carve(p, (size_t)B * N * N * 2);
    __bf16* Xb   = (__bf16*)carve(p, (size_t)B * N * Din * 2);
    __bf16* WfcT = (__bf16*)carve(p, (size_t)H * Din * 2);
    __bf16* Wa1T = (__bf16*)carve(p, (size_t)Adim * H * 2);
    float*  Y0   = (float*)carve(p, (size_t)B * N * H * 4);
    float*  Y1   = (float*)carve(p, (size_t)B * N * H * 4);
    __bf16* Ybf  = (__bf16*)carve(p, (size_t)B * N * H * 2);
    __bf16* Ytbf = (__bf16*)carve(p, (size_t)B * N * H * 2);
    float*  Hh0  = (float*)carve(p, (size_t)B * N * Adim * 4);
    float*  Hh1  = (float*)carve(p, (size_t)B * N * Adim * 4);
    __bf16* Hhbf = (__bf16*)carve(p, (size_t)B * N * Adim * 2);
    __bf16* HhT  = (__bf16*)carve(p, (size_t)B * N * Adim * 2);
    float*  attv = (float*)carve(p, (size_t)B * N * 4);
    float*  attn = (float*)carve(p, (size_t)B * N * 4);
    float*  z    = (float*)carve(p, (size_t)B * H * 4);

    const dim3 tb(32, 8);

    // fp32 -> bf16 operand conversion
    {
        size_t n1 = (size_t)B * N * Din;
        cvt_bf16_kernel<<<(unsigned)((n1 + 255) / 256), 256, 0, stream>>>(X, Xb, n1);
        size_t n2 = (size_t)B * N * N;
        cvt_bf16_kernel<<<(unsigned)((n2 + 255) / 256), 256, 0, stream>>>(adj, adjB, n2);
    }
    // weight transposes (K-contiguous B operands)
    pack_transpose_kernel<<<dim3(H / 32, Din / 32, 1), tb, 0, stream>>>(W_fc, nullptr, WfcT, Din, H);
    pack_transpose_kernel<<<dim3(Adim / 32, H / 32, 1), tb, 0, stream>>>(W_att1, nullptr, Wa1T, H, Adim);

    // FC: Y0 = relu(X @ W_fc + b_fc)   (M=N nodes, N=H, K=Din)
    gemm_wmma_bf16<<<dim3(H / 128, N / 64, B), 256, 0, stream>>>(
        Xb, (long long)N * Din, WfcT, 0, Y0, (long long)N * H,
        nullptr, 0, b_fc, nullptr, 0, N, H, Din, 1);

    // GCN stack: Y = relu(adj @ Y + Y) * mask, 4 layers
    pack_transpose_kernel<<<dim3(H / 32, N / 32, B), tb, 0, stream>>>(Y0, Ybf, Ytbf, N, H);
    float* cur = Y0;
    float* nxt = Y1;
    for (int l = 0; l < 4; ++l) {
        gemm_wmma_bf16<<<dim3(H / 128, N / 64, B), 256, 0, stream>>>(
            adjB, (long long)N * N, Ytbf, (long long)H * N, nxt, (long long)N * H,
            cur, (long long)N * H, nullptr, mask, (long long)N, N, H, N, 1);
        pack_transpose_kernel<<<dim3(H / 32, N / 32, B), tb, 0, stream>>>(nxt, Ybf, Ytbf, N, H);
        float* t = cur; cur = nxt; nxt = t;
    }
    // cur == final Y (f32); Ybf/Ytbf packed from it

    // Attention: Hh0 = Y @ W_att1 + b_att1 (no relu)
    gemm_wmma_bf16<<<dim3(Adim / 128, N / 64, B), 256, 0, stream>>>(
        Ybf, (long long)N * H, Wa1T, 0, Hh0, (long long)N * Adim,
        nullptr, 0, b_att1, nullptr, 0, N, Adim, H, 0);
    pack_transpose_kernel<<<dim3(Adim / 32, N / 32, B), tb, 0, stream>>>(Hh0, Hhbf, HhT, N, Adim);

    // Hh1 = relu(adj @ Hh0 + Hh0) * mask
    gemm_wmma_bf16<<<dim3(Adim / 128, N / 64, B), 256, 0, stream>>>(
        adjB, (long long)N * N, HhT, (long long)Adim * N, Hh1, (long long)N * Adim,
        Hh0, (long long)N * Adim, nullptr, mask, (long long)N, N, Adim, N, 1);

    // att -> softmax -> pooled z -> logits
    att_kernel<<<(B * N) / 256, 256, 0, stream>>>(Hh1, W_att2, b_att2, attv, B * N, Adim);
    softmax_kernel<<<B, 256, 0, stream>>>(attv, attn, N);
    pool_kernel<<<dim3(B, H / 256), 256, 0, stream>>>(cur, attn, z, N, H);
    logits_kernel<<<B, 256, 0, stream>>>(z, W_cls, b_cls, out, H);
}
